// LVSM_41257455845595
// MI455X (gfx1250) — matrix operand order repaired and checked by reading, not compiled, not run on hardware
//
#include <hip/hip_runtime.h>
#include <hip/hip_bf16.h>

typedef __attribute__((ext_vector_type(4)))  __bf16 v4bf;
typedef __attribute__((ext_vector_type(8)))  __bf16 v8bf;
typedef __attribute__((ext_vector_type(16))) __bf16 v16bf;
typedef __attribute__((ext_vector_type(8)))  float  v8f;

#define WMMA_BF16(A,B,C) __builtin_amdgcn_wmma_f32_16x16x32_bf16(false,(A),false,(B),(short)0,(C),false,false)

static __device__ __forceinline__ v16bf cat16(v8bf lo, v8bf hi){
  v16bf r;
#pragma unroll
  for (int i=0;i<8;i++){ r[i]=lo[i]; r[i+8]=hi[i]; }
  return r;
}
static __device__ __forceinline__ v8f zero8(){
  v8f z;
#pragma unroll
  for (int i=0;i<8;i++) z[i]=0.f;
  return z;
}
static __device__ __forceinline__ v4bf pack4(float a, float b, float c, float d){
  v4bf r; r[0]=(__bf16)a; r[1]=(__bf16)b; r[2]=(__bf16)c; r[3]=(__bf16)d; return r;
}

// ---------------- patchify ----------------
__global__ void lvsm_patch_src(const float* __restrict__ img, const float* __restrict__ rays,
                               float* __restrict__ A){
  int idx = blockIdx.x*256 + threadIdx.x;            // 2048*2304 total
  int col = idx % 2304, row = idx / 2304;
  int pw = col & 15, ph = (col>>4) & 15, c = col >> 8;
  int w = row & 15, h = (row>>4) & 15, n = (row>>8) & 3, b = row >> 10;
  int X = w*16 + pw, Y = h*16 + ph;
  float v;
  if (c < 3) v = img [(((long)((b*4+n)*3 +  c   ))*256 + Y)*256 + X];
  else       v = rays[(((long)((b*4+n)*6 + (c-3)))*256 + Y)*256 + X];
  A[idx] = v;
}

__global__ void lvsm_patch_tgt(const float* __restrict__ rays, float* __restrict__ A){
  int idx = blockIdx.x*256 + threadIdx.x;            // 2048*1536 total
  int col = idx % 1536, row = idx / 1536;
  int pw = col & 15, ph = (col>>4) & 15, c = col >> 8;   // c in 0..5
  int w = row & 15, h = (row>>4) & 15, n = (row>>8) & 3, b = row >> 10;
  int X = w*16 + pw, Y = h*16 + ph;
  A[idx] = rays[(((long)((b*4+n)*6 + c))*256 + Y)*256 + X];
}

// ---------------- layernorm (row per block) ----------------
__global__ __launch_bounds__(256) void lvsm_ln(const float* __restrict__ x,
                                               const float* __restrict__ s,
                                               const float* __restrict__ b,
                                               float* __restrict__ y){
  const int row = blockIdx.x, tid = threadIdx.x;
  __shared__ float sbuf[256];
  const float* xr = x + (long)row*768;
  float v0 = xr[tid], v1 = xr[tid+256], v2 = xr[tid+512];
  sbuf[tid] = v0+v1+v2; __syncthreads();
  for (int o=128;o>0;o>>=1){ if (tid<o) sbuf[tid]+=sbuf[tid+o]; __syncthreads(); }
  float mean = sbuf[0]*(1.f/768.f); __syncthreads();
  sbuf[tid] = v0*v0+v1*v1+v2*v2; __syncthreads();
  for (int o=128;o>0;o>>=1){ if (tid<o) sbuf[tid]+=sbuf[tid+o]; __syncthreads(); }
  float var = sbuf[0]*(1.f/768.f) - mean*mean;
  float inv = rsqrtf(var + 1e-5f);
  float* yr = y + (long)row*768;
  yr[tid]     = (v0-mean)*inv*s[tid]     + b[tid];
  yr[tid+256] = (v1-mean)*inv*s[tid+256] + b[tid+256];
  yr[tid+512] = (v2-mean)*inv*s[tid+512] + b[tid+512];
}

// ---------------- generic WMMA GEMM: out = epi(Aremap @ W + bias [+res]) ----------------
// Block tile 128x128, K-step 32, 8 waves each 32x64, double-buffered LDS.
// Row remap (power-of-two segments): phys_row(m) = (m>>shift)*ldseg + base + (m & ((1<<shift)-1))
__global__ __launch_bounds__(256) void lvsm_gemm(
    const float* __restrict__ A, int K,
    const float* __restrict__ W, const float* __restrict__ bias,
    const float* __restrict__ res, float* __restrict__ out, int N,
    int aShift, int ldsegA, int baseA,
    int oShift, int ldsegO, int baseO, int epi){
  const int m0 = blockIdx.x*128, n0 = blockIdx.y*128;
  const int tid = threadIdx.x, lane = tid & 31, wid = tid >> 5;
  const int g = (lane>>4) & 1, l16 = lane & 15;
  const int wm = wid & 3, wn = wid >> 2;
  const int aMask = (1<<aShift)-1, oMask = (1<<oShift)-1;
  __shared__ __bf16 As[2][128][40];   // [buf][m][k], stride 40 (80B)
  __shared__ __bf16 Bs[2][128][40];   // [buf][n][k] (W transposed in LDS)

  // staging coords hoisted out of the K loop
  long arowB[4];
  const float* wcol[4];
  int aRow[4], aKg[4], bN[4], bKg[4];
#pragma unroll
  for (int i=0;i<4;i++){
    int lin = tid + 256*i;
    aRow[i] = lin >> 3; aKg[i] = lin & 7;            // A: 128 rows x 8 kgroups
    int gm = m0 + aRow[i];
    arowB[i] = ((long)(gm>>aShift)*ldsegA + baseA + (gm & aMask))*(long)K;
    bN[i] = lin & 127; bKg[i] = lin >> 7;            // B: 128 n x 8 kgroups(4k each)
    wcol[i] = W + (long)bKg[i]*4*N + n0 + bN[i];
  }

  float4 fa[4];
  float  fw[4][4];
  auto loadTile = [&](int k0){
#pragma unroll
    for (int i=0;i<4;i++){
      fa[i] = *(const float4*)&A[arowB[i] + k0 + aKg[i]*4];
      const float* wp = wcol[i] + (long)k0*N;
#pragma unroll
      for (int j=0;j<4;j++) fw[i][j] = wp[(long)j*N];  // coalesced along n across lanes
    }
  };
  auto storeTile = [&](int buf){
#pragma unroll
    for (int i=0;i<4;i++){
      *(v4bf*)&As[buf][aRow[i]][aKg[i]*4] = pack4(fa[i].x, fa[i].y, fa[i].z, fa[i].w);
      *(v4bf*)&Bs[buf][bN[i]][bKg[i]*4]   = pack4(fw[i][0], fw[i][1], fw[i][2], fw[i][3]);
    }
  };

  v8f acc[2][4];
#pragma unroll
  for (int i=0;i<2;i++)
#pragma unroll
    for (int j=0;j<4;j++) acc[i][j] = zero8();

  loadTile(0);
  storeTile(0);
  const int nk = K >> 5;
  for (int step=0; step<nk; step++){
    const int cur = step & 1;
    if (step+1 < nk) loadTile((step+1) << 5);        // prefetch next tile into regs
    __syncthreads();                                 // buf[cur] ready for all waves
    v16bf af[2];
#pragma unroll
    for (int mt=0;mt<2;mt++){
      int m = wm*32 + mt*16 + l16;
      af[mt] = cat16(*(const v8bf*)&As[cur][m][8*g], *(const v8bf*)&As[cur][m][8*g+16]);
    }
#pragma unroll
    for (int nt=0;nt<4;nt++){
      int n = wn*64 + nt*16 + l16;
      v16bf bfm = cat16(*(const v8bf*)&Bs[cur][n][8*g], *(const v8bf*)&Bs[cur][n][8*g+16]);
#pragma unroll
      for (int mt=0;mt<2;mt++)
        acc[mt][nt] = WMMA_BF16(af[mt], bfm, acc[mt][nt]);
    }
    if (step+1 < nk) storeTile(cur ^ 1);             // fill other buffer for next step
  }
  // epilogue: C layout -> M = r + 8*g, N = l16 within each 16x16 tile
#pragma unroll
  for (int mt=0;mt<2;mt++){
#pragma unroll
    for (int nt=0;nt<4;nt++){
#pragma unroll
      for (int r=0;r<8;r++){
        int gm = m0 + wm*32 + mt*16 + r + 8*g;
        int gn = n0 + wn*64 + nt*16 + l16;
        long orow = (long)(gm>>oShift)*ldsegO + baseO + (gm & oMask);
        float v = acc[mt][nt][r] + bias[gn];
        if (epi == 1){
          v += res[orow*N + gn];
        } else if (epi == 2){
          float u = v;
          float cc = 0.7978845608028654f*(u + 0.044715f*u*u*u);
          v = 0.5f*u*(1.0f + tanhf(cc));
        } else if (epi == 3){
          v = 1.0f/(1.0f + __expf(-v));
        }
        out[orow*N + gn] = v;
      }
    }
  }
}

// ---------------- attention: flash-style, mask by key-range iteration ----------------
// qkv: [B, 2048, 3*768] fp32, out: [B, 2048, 768] fp32.
// grid: (T/128=16, NH=12, B=2), 256 threads = 8 waves, 16 queries per wave.
__global__ __launch_bounds__(256) void lvsm_attn(const float* __restrict__ qkv,
                                                 float* __restrict__ out){
  const int qb = blockIdx.x * 128;
  const int head = blockIdx.y;
  const int b = blockIdx.z;
  const int tid = threadIdx.x, lane = tid & 31, wid = tid >> 5;
  const int g = (lane>>4) & 1, l16 = lane & 15;
  __shared__ __bf16 Qs[128][72];    // [q][d]
  __shared__ __bf16 Ks[32][72];     // [key][d]  (B for scores)
  __shared__ __bf16 Vt[64][40];     // [d][key]  (B for P@V)
  __shared__ __bf16 Ps[8][16][40];  // per-wave P tile [M][key]
  const float scale = 0.125f;       // 1/sqrt(64)
  const long qkvb = (long)b*2048*2304;

  // stage Q tile (128 x 64) as bf16
#pragma unroll
  for (int i=0;i<8;i++){
    int lin = tid + 256*i;                  // 2048 float4
    int qr = lin >> 4, dg = lin & 15;
    const float4 f = *(const float4*)&qkv[qkvb + (long)(qb+qr)*2304 + head*64 + dg*4];
    *(v4bf*)&Qs[qr][dg*4] = pack4(f.x, f.y, f.z, f.w);
  }
  __syncthreads();
  v16bf aq[2];
#pragma unroll
  for (int s=0;s<2;s++){
    int m = wid*16 + l16;
    aq[s] = cat16(*(const v8bf*)&Qs[m][32*s + 8*g],
                  *(const v8bf*)&Qs[m][32*s + 8*g + 16]);
  }
  float mrun[8], lrun[8];
  v8f accO[4];
#pragma unroll
  for (int r=0;r<8;r++){ mrun[r] = -3.0e38f; lrun[r] = 0.f; }
#pragma unroll
  for (int n=0;n<4;n++) accO[n] = zero8();

  const int view = (qb >= 1024) ? ((qb - 1024) >> 8) : 0;
  const int nsteps = (qb < 1024) ? 32 : 40;  // 1024 src keys (+256 own-view keys for tgt)

  for (int st=0; st<nsteps; st++){
    const int kb = (st < 32) ? st*32 : 1024 + view*256 + (st-32)*32;
    __syncthreads();
    // stage K (natural) and V (transposed), 32 keys x 64 dims, packed b64 LDS stores
#pragma unroll
    for (int i=0;i<2;i++){
      int lin = tid + 256*i;                // K: 512 float4
      int kr = lin >> 4, dg = lin & 15;
      const float4 kf = *(const float4*)&qkv[qkvb + (long)(kb+kr)*2304 + 768 + head*64 + dg*4];
      *(v4bf*)&Ks[kr][dg*4] = pack4(kf.x, kf.y, kf.z, kf.w);
      int d = lin & 63, kg = lin >> 6;      // V: 64 d x 8 keygroups(4 keys each)
      const float* vp = &qkv[qkvb + (long)(kb+kg*4)*2304 + 1536 + head*64 + d];
      *(v4bf*)&Vt[d][kg*4] = pack4(vp[0], vp[2304], vp[4608], vp[6912]);
    }
    __syncthreads();
    // scores: Q(16x64) @ K^T -> two 16x16 tiles (keys 0-15, 16-31)
    v8f s0 = zero8(), s1 = zero8();
#pragma unroll
    for (int s=0;s<2;s++){
      v16bf bk0 = cat16(*(const v8bf*)&Ks[l16][32*s + 8*g],
                        *(const v8bf*)&Ks[l16][32*s + 8*g + 16]);
      s0 = WMMA_BF16(aq[s], bk0, s0);
      v16bf bk1 = cat16(*(const v8bf*)&Ks[l16+16][32*s + 8*g],
                        *(const v8bf*)&Ks[l16+16][32*s + 8*g + 16]);
      s1 = WMMA_BF16(aq[s], bk1, s1);
    }
    // online softmax per row (row = r + 8*g, replicated across 16-lane half)
    float p0[8], p1[8];
#pragma unroll
    for (int r=0;r<8;r++){
      float t = fmaxf(s0[r], s1[r]);
#pragma unroll
      for (int m=1;m<16;m<<=1) t = fmaxf(t, __shfl_xor(t, m, 32));
      float mn = fmaxf(mrun[r], scale*t);
      float corr = __expf(mrun[r] - mn);
      float a0 = __expf(scale*s0[r] - mn);
      float a1 = __expf(scale*s1[r] - mn);
      float rs = a0 + a1;
#pragma unroll
      for (int m=1;m<16;m<<=1) rs += __shfl_xor(rs, m, 32);
      lrun[r] = lrun[r]*corr + rs;
      mrun[r] = mn;
      p0[r] = a0; p1[r] = a1;
#pragma unroll
      for (int n=0;n<4;n++) accO[n][r] = accO[n][r]*corr;
    }
    // C-layout -> A-layout via per-wave LDS round trip
#pragma unroll
    for (int r=0;r<8;r++){
      Ps[wid][r + 8*g][l16]      = (__bf16)p0[r];
      Ps[wid][r + 8*g][l16 + 16] = (__bf16)p1[r];
    }
    __syncthreads();
    v16bf pa = cat16(*(const v8bf*)&Ps[wid][l16][8*g],
                     *(const v8bf*)&Ps[wid][l16][8*g + 16]);
#pragma unroll
    for (int n=0;n<4;n++){
      v16bf vb = cat16(*(const v8bf*)&Vt[n*16 + l16][8*g],
                       *(const v8bf*)&Vt[n*16 + l16][8*g + 16]);
      accO[n] = WMMA_BF16(pa, vb, accO[n]);
    }
  }
  const long ob = (long)b*2048*768;
#pragma unroll
  for (int r=0;r<8;r++){
    float invl = 1.0f/lrun[r];
    int q = qb + wid*16 + r + 8*g;
#pragma unroll
    for (int n=0;n<4;n++)
      out[ob + (long)q*768 + head*64 + n*16 + l16] = accO[n][r]*invl;
  }
}

// ---------------- unpatchify ----------------
__global__ void lvsm_unpatch(const float* __restrict__ P, float* __restrict__ out){
  int idx = blockIdx.x*256 + threadIdx.x;            // 2048*768
  int col = idx % 768, row = idx / 768;
  int pw = col & 15, ph = (col>>4) & 15, c = col >> 8;
  int w = row & 15, h = (row>>4) & 15, n = (row>>8) & 3, b = row >> 10;
  out[(((long)((b*4+n)*3 + c)*256 + h*16+ph)*256) + w*16+pw] = P[idx];
}

extern "C" void kernel_launch(void* const* d_in, const int* in_sizes, int n_in,
                              void* d_out, int out_size, void* d_ws, size_t ws_size,
                              hipStream_t stream){
  const float* src_img  = (const float*)d_in[0];
  const float* src_rays = (const float*)d_in[1];
  const float* tgt_rays = (const float*)d_in[2];
  const float* W_in  = (const float*)d_in[3];
  const float* b_in  = (const float*)d_in[4];
  const float* W_tgt = (const float*)d_in[5];
  const float* b_tgt = (const float*)d_in[6];
  const float* ln1_s = (const float*)d_in[7];
  const float* ln1_b = (const float*)d_in[8];
  const float* Wqkv  = (const float*)d_in[9];
  const float* bqkv  = (const float*)d_in[10];
  const float* Wo    = (const float*)d_in[11];
  const float* bo    = (const float*)d_in[12];
  const float* ln2_s = (const float*)d_in[13];
  const float* ln2_b = (const float*)d_in[14];
  const float* W1    = (const float*)d_in[15];
  const float* b1    = (const float*)d_in[16];
  const float* W2    = (const float*)d_in[17];
  const float* b2    = (const float*)d_in[18];
  const float* lnf_s = (const float*)d_in[19];
  const float* lnf_b = (const float*)d_in[20];
  const float* W_out = (const float*)d_in[21];
  const float* b_out = (const float*)d_in[22];

  float* x    = (float*)d_ws;                    // [2,2048,768]
  float* hdn  = x    + (size_t)4096*768;
  float* qkv  = hdn  + (size_t)4096*768;         // [2,2048,2304]
  float* attO = qkv  + (size_t)4096*2304;        // [2,2048,768]
  float* hbuf = attO + (size_t)4096*768;         // [2,2048,3072]
  float* Asrc = qkv;                             // reuse qkv region pre-layers
  float* Atgt = qkv  + (size_t)2048*2304;
  float* outp = attO;                            // reuse post-layers

  // patchify
  lvsm_patch_src<<<(2048*2304)/256, 256, 0, stream>>>(src_img, src_rays, Asrc);
  lvsm_patch_tgt<<<(2048*1536)/256, 256, 0, stream>>>(tgt_rays, Atgt);
  // token embeddings (scatter into x with src/tgt interleave per batch)
  lvsm_gemm<<<dim3(16, 6), 256, 0, stream>>>(Asrc, 2304, W_in, b_in, nullptr, x, 768,
      11,2048,0,  10,2048,0,    0);
  lvsm_gemm<<<dim3(16, 6), 256, 0, stream>>>(Atgt, 1536, W_tgt, b_tgt, nullptr, x, 768,
      11,2048,0,  10,2048,1024, 0);

  for (int i=0;i<6;i++){
    lvsm_ln<<<4096, 256, 0, stream>>>(x, ln1_s + i*768, ln1_b + i*768, hdn);
    lvsm_gemm<<<dim3(32, 18), 256, 0, stream>>>(hdn, 768,
        Wqkv + (size_t)i*768*2304, bqkv + i*2304, nullptr, qkv, 2304,
        12,4096,0, 12,4096,0, 0);
    lvsm_attn<<<dim3(16, 12, 2), 256, 0, stream>>>(qkv, attO);
    lvsm_gemm<<<dim3(32, 6), 256, 0, stream>>>(attO, 768,
        Wo + (size_t)i*768*768, bo + i*768, x, x, 768,
        12,4096,0, 12,4096,0, 1);
    lvsm_ln<<<4096, 256, 0, stream>>>(x, ln2_s + i*768, ln2_b + i*768, hdn);
    lvsm_gemm<<<dim3(32, 24), 256, 0, stream>>>(hdn, 768,
        W1 + (size_t)i*768*3072, b1 + i*3072, nullptr, hbuf, 3072,
        12,4096,0, 12,4096,0, 2);
    lvsm_gemm<<<dim3(32, 6), 256, 0, stream>>>(hbuf, 3072,
        W2 + (size_t)i*3072*768, b2 + i*768, x, x, 768,
        12,4096,0, 12,4096,0, 1);
  }

  lvsm_ln<<<4096, 256, 0, stream>>>(x, lnf_s, lnf_b, hdn);
  // out projection over tgt tokens only (gather rows b*2048+1024..2047), sigmoid
  lvsm_gemm<<<dim3(16, 6), 256, 0, stream>>>(hdn, 768, W_out, b_out, nullptr, outp, 768,
      10,2048,1024, 11,2048,0, 3);
  lvsm_unpatch<<<(2048*768)/256, 256, 0, stream>>>(outp, (float*)d_out);
}
